// MultiHeadAttention_24541443129593
// MI455X (gfx1250) — matrix operand order repaired and checked by reading, not compiled, not run on hardware
//
#include <hip/hip_runtime.h>
#include <hip/hip_bf16.h>
#include <math.h>

// ---------------- problem constants ----------------
constexpr int BB  = 4;
constexpr int HH  = 16;
constexpr int SS  = 1024;
constexpr int MEMN = 1024;
constexpr int KVN = 2048;     // SS + MEMN
constexpr int DD  = 64;
constexpr int HID = 1024;

constexpr int QT  = 64;       // q rows per workgroup in attention
constexpr int KVT = 64;       // kv tile in flash loop
constexpr int BDROWS = 65;    // 64 rows + 1 extra for rel-shift wrap

// ---------------- vector types ----------------
typedef __attribute__((ext_vector_type(16))) __bf16 v16bf;
typedef __attribute__((ext_vector_type(8)))  __bf16 v8bf;
typedef __attribute__((ext_vector_type(8)))  float  v8f;
typedef __attribute__((ext_vector_type(4)))  int    v4i_t;

// ---------------- workspace layout (in __bf16 elements) ----------------
constexpr size_t WSZ     = (size_t)HID * HID;                    // 1M elems per weight
constexpr size_t OFF_WQ  = 0;
constexpr size_t OFF_WK  = OFF_WQ  + WSZ;
constexpr size_t OFF_WV  = OFF_WK  + WSZ;
constexpr size_t OFF_WR  = OFF_WV  + WSZ;
constexpr size_t OFF_WFC = OFF_WR  + WSZ;
constexpr size_t OFF_XB  = OFF_WFC + WSZ;                        // x       (B*S*HID)
constexpr size_t OFF_XE  = OFF_XB  + (size_t)BB * SS  * HID;     // x_extra (B*KV*HID)
constexpr size_t OFF_REL = OFF_XE  + (size_t)BB * KVN * HID;     // rel     (KV*HID)
constexpr size_t OFF_QU  = OFF_REL + (size_t)KVN * HID;          // q+u     (B,H,S,D)
constexpr size_t OFF_QV  = OFF_QU  + (size_t)BB * HH * SS * DD;  // q+v_rel
constexpr size_t OFF_K   = OFF_QV  + (size_t)BB * HH * SS * DD;  // k       (B,H,KV,D)
constexpr size_t OFF_VT  = OFF_K   + (size_t)BB * HH * KVN * DD; // v^T     (B,H,D,KV)
constexpr size_t OFF_R   = OFF_VT  + (size_t)BB * HH * KVN * DD; // r       (H,KV,D)
constexpr size_t OFF_CTX = OFF_R   + (size_t)HH * KVN * DD;      // attn ctx (B,S,HID)

// ---------------- CDNA5 async global->LDS copy (ASYNCcnt path) ----------------
__device__ __forceinline__ void cp_async16(const __bf16* g, __bf16* l) {
#if __has_builtin(__builtin_amdgcn_global_load_async_to_lds_b128)
  __builtin_amdgcn_global_load_async_to_lds_b128((v4i_t*)g, (v4i_t*)l, 0, 0);
#else
  *(v8bf*)l = *(const v8bf*)g;
#endif
}

__device__ __forceinline__ void wait_async0() {
#if __has_builtin(__builtin_amdgcn_s_wait_asynccnt)
  __builtin_amdgcn_s_wait_asynccnt(0);
#else
  asm volatile("s_wait_asynccnt 0" ::: "memory");
#endif
}

// ---------------- WMMA fragment loaders (per ISA VGPR layouts) ----------------
// A matrix 16x32 bf16: lanes 0-15 -> row = lane, K 0..7 + 16..23
//                      lanes 16-31 -> row = lane-16, K 8..15 + 24..31
__device__ __forceinline__ v16bf ld_a(const __bf16* tile, int ld, int lane, int k0) {
  const __bf16* p = tile + (size_t)(lane & 15) * ld + k0 + ((lane >> 4) << 3);
  v8bf lo = *(const v8bf*)p;
  v8bf hi = *(const v8bf*)(p + 16);
  return __builtin_shufflevector(lo, hi, 0,1,2,3,4,5,6,7,8,9,10,11,12,13,14,15);
}

// B matrix 32x16 bf16: lanes 0-15 -> col n = lane, K 0..7 + 16..23
//                      lanes 16-31 -> col n = lane-16, K 8..15 + 24..31
// Source element B[k][n] lives at tile[n*ld + k]  (n-major rows, contiguous in k)
__device__ __forceinline__ v16bf ld_b(const __bf16* tile, int ld, int lane, int k0) {
  const __bf16* p = tile + (size_t)(lane & 15) * ld + k0 + ((lane >> 4) << 3);
  v8bf lo = *(const v8bf*)p;
  v8bf hi = *(const v8bf*)(p + 16);
  return __builtin_shufflevector(lo, hi, 0,1,2,3,4,5,6,7,8,9,10,11,12,13,14,15);
}

__device__ __forceinline__ v8f wmma_bf16(v16bf a, v16bf b, v8f c) {
  return __builtin_amdgcn_wmma_f32_16x16x32_bf16(false, a, false, b, (short)0, c,
                                                 false, false);
}

__device__ __forceinline__ float redmax16(float v) {
  v = fmaxf(v, __shfl_xor(v, 1, 32));
  v = fmaxf(v, __shfl_xor(v, 2, 32));
  v = fmaxf(v, __shfl_xor(v, 4, 32));
  v = fmaxf(v, __shfl_xor(v, 8, 32));
  return v;
}
__device__ __forceinline__ float redsum16(float v) {
  v += __shfl_xor(v, 1, 32);
  v += __shfl_xor(v, 2, 32);
  v += __shfl_xor(v, 4, 32);
  v += __shfl_xor(v, 8, 32);
  return v;
}

// ---------------- kernels ----------------
__global__ void __launch_bounds__(256)
k_cvt(const float* __restrict__ s, __bf16* __restrict__ d, int n) {
  int i = (blockIdx.x * blockDim.x + threadIdx.x) * 4;
  if (i + 4 <= n) {
    float4 f = *(const float4*)(s + i);
    d[i + 0] = (__bf16)f.x;
    d[i + 1] = (__bf16)f.y;
    d[i + 2] = (__bf16)f.z;
    d[i + 3] = (__bf16)f.w;
  }
}

__global__ void __launch_bounds__(256)
k_build_xe(const float* __restrict__ pkv, const float* __restrict__ x,
           __bf16* __restrict__ xe) {
  int e = (blockIdx.x * blockDim.x + threadIdx.x) * 4;
  const int total = BB * KVN * HID;
  if (e + 4 > total) return;
  int b   = e / (KVN * HID);
  int rem = e - b * (KVN * HID);
  int row = rem / HID;
  int c   = rem - row * HID;
  const float* src = (row < MEMN) ? (pkv + ((size_t)b * MEMN + row) * HID + c)
                                  : (x   + ((size_t)b * SS + (row - MEMN)) * HID + c);
  float4 f = *(const float4*)src;
  xe[e + 0] = (__bf16)f.x;
  xe[e + 1] = (__bf16)f.y;
  xe[e + 2] = (__bf16)f.z;
  xe[e + 3] = (__bf16)f.w;
}

// Projection GEMMs: one wave per 16x16 output tile, K = HID.
// All 8 waves in a block share the same 16-row A tile -> stage it once in LDS
// via the CDNA5 async global->LDS engine.
// job 0: q = x@Wq^T  -> qu = q+u, qv = q+v_rel, layout (B,H,S,D)
// job 1: k = xe@Wk^T -> (B,H,KV,D)
// job 2: v = xe@Wv^T -> transposed (B,H,D,KV)
// job 3: r = rel@Wr^T-> (H,KV,D)
constexpr int NT16 = HID / 16;                      // 64
constexpr int TQ = (BB * SS / 16) * NT16;           // 16384
constexpr int TK = (BB * KVN / 16) * NT16;          // 32768
constexpr int TR = (KVN / 16) * NT16;               // 8192
constexpr int TPROJ = TQ + 2 * TK + TR;             // 90112

__global__ void __launch_bounds__(256)
k_proj(const __bf16* __restrict__ xb, const __bf16* __restrict__ xe,
       const __bf16* __restrict__ relb,
       const __bf16* __restrict__ wq, const __bf16* __restrict__ wk,
       const __bf16* __restrict__ wv, const __bf16* __restrict__ wr,
       const float* __restrict__ u, const float* __restrict__ vrel,
       __bf16* __restrict__ qu, __bf16* __restrict__ qv,
       __bf16* __restrict__ kk, __bf16* __restrict__ vt,
       __bf16* __restrict__ rr) {
  __shared__ __bf16 a_lds[16 * HID];                // 32 KB, shared by 8 waves
  int wt = blockIdx.x * 8 + (threadIdx.x >> 5);
  int lane = threadIdx.x & 31;

  int job, r;
  const __bf16 *A, *W;
  if (wt < TQ)                { job = 0; r = wt;                A = xb;   W = wq; }
  else if (wt < TQ + TK)      { job = 1; r = wt - TQ;           A = xe;   W = wk; }
  else if (wt < TQ + 2 * TK)  { job = 2; r = wt - TQ - TK;      A = xe;   W = wv; }
  else                        { job = 3; r = wt - TQ - 2 * TK;  A = relb; W = wr; }

  int mt = r / NT16, nt = r % NT16;                 // mt identical across the block
  const __bf16* at = A + (size_t)mt * 16 * HID;
  const __bf16* bt = W + (size_t)nt * 16 * HID;

  for (int i = threadIdx.x; i < 16 * HID / 8; i += 256)
    cp_async16(at + (size_t)i * 8, a_lds + (size_t)i * 8);
  wait_async0();
  __syncthreads();

  v8f c = {};
  for (int k0 = 0; k0 < HID; k0 += 32) {
    v16bf a = ld_a(a_lds, HID, lane, k0);
    v16bf b = ld_b(bt, HID, lane, k0);
    c = wmma_bf16(a, b, c);
  }

  int hi = lane >> 4;
  int n  = nt * 16 + (lane & 15);
  int h  = n >> 6, d = n & 63;
#pragma unroll
  for (int g = 0; g < 8; ++g) {
    int   m = mt * 16 + g + 8 * hi;
    float v = c[g];
    if (job == 0) {
      int b_ = m >> 10, s = m & (SS - 1);
      size_t o = ((size_t)(b_ * HH + h) * SS + s) * DD + d;
      qu[o] = (__bf16)(v + u[h * DD + d]);
      qv[o] = (__bf16)(v + vrel[h * DD + d]);
    } else if (job == 1) {
      int b_ = m >> 11, kv = m & (KVN - 1);
      kk[((size_t)(b_ * HH + h) * KVN + kv) * DD + d] = (__bf16)v;
    } else if (job == 2) {
      int b_ = m >> 11, kv = m & (KVN - 1);
      vt[((size_t)(b_ * HH + h) * DD + d) * KVN + kv] = (__bf16)v;
    } else {
      rr[((size_t)h * KVN + m) * DD + d] = (__bf16)v;
    }
  }
}

// Fused rel-shift flash attention. One WG = (b, h, 64-row q tile), 4 waves.
__global__ void __launch_bounds__(128)
k_attn(const __bf16* __restrict__ qu, const __bf16* __restrict__ qv,
       const __bf16* __restrict__ kk, const __bf16* __restrict__ vt,
       const __bf16* __restrict__ rr, const int* __restrict__ mask,
       __bf16* __restrict__ ctx) {
  extern __shared__ char smem_raw[];
  __bf16* bd  = (__bf16*)smem_raw;          // [BDROWS][KVN]
  __bf16* qus = bd  + (size_t)BDROWS * KVN; // [QT][DD]
  __bf16* qvs = qus + QT * DD;              // [QT][DD]
  __bf16* pst = qvs + QT * DD;              // [4 waves][16][KVT]

  int blk = blockIdx.x;
  int q0  = (blk & 15) * QT;
  int h   = (blk >> 4) & 15;
  int b   = blk >> 8;
  int tid = threadIdx.x, lane = tid & 31, w = tid >> 5;

  const __bf16* qu_bh = qu + (size_t)(b * HH + h) * SS * DD;
  const __bf16* qv_bh = qv + (size_t)(b * HH + h) * SS * DD;
  const __bf16* k_bh  = kk + (size_t)(b * HH + h) * KVN * DD;
  const __bf16* v_bh  = vt + (size_t)(b * HH + h) * DD * KVN;
  const __bf16* r_h   = rr + (size_t)h * KVN * DD;

  // stage this tile's qu/qv rows (64x64 each) into LDS via async engine
  for (int i = tid; i < QT * DD / 8; i += 128) {
    cp_async16(qu_bh + (size_t)q0 * DD + (size_t)i * 8, qus + (size_t)i * 8);
    cp_async16(qv_bh + (size_t)q0 * DD + (size_t)i * 8, qvs + (size_t)i * 8);
  }
  wait_async0();
  __syncthreads();

  // ---- phase A: BDraw = (q+v_rel) @ r^T for rows q0..q0+64, full KV, into LDS ----
  for (int c = w; c < 5; c += 4) {
    bool extra = (c == 4);
    if (extra && q0 + QT >= SS) {
      for (int j = lane; j < KVN; j += 32) bd[(size_t)64 * KVN + j] = (__bf16)0.0f;
      continue;
    }
    const __bf16* atile = extra ? (qv_bh + (size_t)(q0 + QT) * DD)
                                : (qvs + (size_t)c * 16 * DD);
    for (int j0 = 0; j0 < KVN; j0 += 16) {
      v8f acc = {};
#pragma unroll
      for (int k0 = 0; k0 < DD; k0 += 32) {
        v16bf a = ld_a(atile, DD, lane, k0);
        v16bf bm = ld_b(r_h + (size_t)j0 * DD, DD, lane, k0);
        acc = wmma_bf16(a, bm, acc);
      }
      if (!extra) {
        int col = j0 + (lane & 15);
        int rbase = c * 16 + 8 * (lane >> 4);
#pragma unroll
        for (int g = 0; g < 8; ++g)
          bd[(size_t)(rbase + g) * KVN + col] = (__bf16)acc[g];
      } else if (lane < 16) {
        bd[(size_t)64 * KVN + j0 + lane] = (__bf16)acc[0];  // M=0 row only
      }
    }
  }
  __syncthreads();

  // ---- phase B: flash attention over KV tiles ----
  const float scale = 0.125f;  // 1/sqrt(64)
  float Mx[8], Ls[8];
  v8f   O[4] = {};
#pragma unroll
  for (int g = 0; g < 8; ++g) { Mx[g] = -INFINITY; Ls[g] = 0.0f; }

  int hi = lane >> 4, ncol = lane & 15;
  __bf16* pw = pst + (size_t)w * 16 * KVT;

  for (int kv0 = 0; kv0 < KVN; kv0 += KVT) {
    if (kv0 + KVT < KVN)
      __builtin_prefetch(k_bh + (size_t)(kv0 + KVT) * DD, 0, 1);

    // S_AC = (q+u) @ k^T for this wave's 16 rows x KVT cols
    v8f Sv[4];
#pragma unroll
    for (int t = 0; t < 4; ++t) {
      v8f acc = {};
#pragma unroll
      for (int k0 = 0; k0 < DD; k0 += 32) {
        v16bf a  = ld_a(qus + (size_t)(16 * w) * DD, DD, lane, k0);
        v16bf bm = ld_b(k_bh + (size_t)(kv0 + 16 * t) * DD, DD, lane, k0);
        acc = wmma_bf16(a, bm, acc);
      }
      Sv[t] = acc;
    }

    // add rel-shifted BD, mask, scale; online softmax update
#pragma unroll
    for (int g = 0; g < 8; ++g) {
      int i_loc  = 16 * w + g + 8 * hi;
      int i_glob = q0 + i_loc;
      float vmax = -INFINITY;
#pragma unroll
      for (int t = 0; t < 4; ++t) {
        int j  = kv0 + 16 * t + ncol;
        int js = j + 1023 - i_glob;
        float bdv;
        if (js <= KVN - 1)            bdv = (float)bd[(size_t)i_loc * KVN + js];
        else if (j == i_glob + 1025)  bdv = 0.0f;
        else                          bdv = (float)bd[(size_t)(i_loc + 1) * KVN +
                                                      (j - i_glob - 1026)];
        float s = (Sv[t][g] + bdv) * scale;
        if (mask[(size_t)i_glob * KVN + j] == 1) s = -INFINITY;
        Sv[t][g] = s;
        vmax = fmaxf(vmax, s);
      }
      vmax = redmax16(vmax);

      float mnew = fmaxf(Mx[g], vmax);
      float corr = (Mx[g] == -INFINITY) ? 0.0f : __expf(Mx[g] - mnew);
      float lsum = 0.0f;
#pragma unroll
      for (int t = 0; t < 4; ++t) {
        float p = (mnew == -INFINITY) ? 0.0f : __expf(Sv[t][g] - mnew);
        lsum += p;
        pw[(size_t)(g + 8 * hi) * KVT + 16 * t + ncol] = (__bf16)p;
      }
      lsum  = redsum16(lsum);
      Ls[g] = Ls[g] * corr + lsum;
      Mx[g] = mnew;
#pragma unroll
      for (int t = 0; t < 4; ++t) O[t][g] *= corr;
    }

    // O += P @ V   (A = P from per-wave LDS, B = v^T tiles)
#pragma unroll
    for (int t = 0; t < 4; ++t) {
      v8f acc = O[t];
#pragma unroll
      for (int k0 = 0; k0 < KVT; k0 += 32) {
        v16bf a  = ld_a(pw, KVT, lane, k0);
        v16bf bm = ld_b(v_bh + (size_t)(16 * t) * KVN + kv0, KVN, lane, k0);
        acc = wmma_bf16(a, bm, acc);
      }
      O[t] = acc;
    }
  }

  // normalize + store context (merged-head layout) as bf16
#pragma unroll
  for (int g = 0; g < 8; ++g) {
    float inv = (Ls[g] > 0.0f) ? (1.0f / Ls[g]) : 0.0f;
    int s = q0 + 16 * w + g + 8 * hi;
#pragma unroll
    for (int t = 0; t < 4; ++t) {
      int dcol = 16 * t + ncol;
      ctx[((size_t)b * SS + s) * HID + h * DD + dcol] = (__bf16)(O[t][g] * inv);
    }
  }
}

// Final projection: out = ctx @ Wfc^T + bfc + x  (fp32 output)
constexpr int TFC = (BB * SS / 16) * NT16;  // 16384

__global__ void __launch_bounds__(256)
k_fc(const __bf16* __restrict__ ctx, const __bf16* __restrict__ wfc,
     const float* __restrict__ bfc, const float* __restrict__ x,
     float* __restrict__ out) {
  __shared__ __bf16 a_lds[16 * HID];
  int wt = blockIdx.x * 8 + (threadIdx.x >> 5);
  int lane = threadIdx.x & 31;
  int mt = wt / NT16, nt = wt % NT16;               // mt identical across block
  const __bf16* at = ctx + (size_t)mt * 16 * HID;
  const __bf16* bt = wfc + (size_t)nt * 16 * HID;

  for (int i = threadIdx.x; i < 16 * HID / 8; i += 256)
    cp_async16(at + (size_t)i * 8, a_lds + (size_t)i * 8);
  wait_async0();
  __syncthreads();

  v8f c = {};
  for (int k0 = 0; k0 < HID; k0 += 32) {
    v16bf a = ld_a(a_lds, HID, lane, k0);
    v16bf b = ld_b(bt, HID, lane, k0);
    c = wmma_bf16(a, b, c);
  }
  int hi = lane >> 4, n = nt * 16 + (lane & 15);
#pragma unroll
  for (int g = 0; g < 8; ++g) {
    int m = mt * 16 + g + 8 * hi;
    out[(size_t)m * HID + n] = c[g] + bfc[n] + x[(size_t)m * HID + n];
  }
}

// ---------------- launch ----------------
extern "C" void kernel_launch(void* const* d_in, const int* in_sizes, int n_in,
                              void* d_out, int out_size, void* d_ws, size_t ws_size,
                              hipStream_t stream) {
  (void)in_sizes; (void)n_in; (void)out_size; (void)ws_size;
  const float* x    = (const float*)d_in[0];
  const float* u    = (const float*)d_in[1];
  const float* vrel = (const float*)d_in[2];
  const float* rel  = (const float*)d_in[3];
  const int*   mask = (const int*)d_in[4];
  const float* pkv  = (const float*)d_in[5];
  const float* Wq   = (const float*)d_in[6];
  const float* Wk   = (const float*)d_in[7];
  const float* Wv   = (const float*)d_in[8];
  const float* Wr   = (const float*)d_in[9];
  const float* Wfc  = (const float*)d_in[10];
  const float* bfc  = (const float*)d_in[11];
  float* out = (float*)d_out;

  __bf16* wsb = (__bf16*)d_ws;
  __bf16* wq   = wsb + OFF_WQ;
  __bf16* wk   = wsb + OFF_WK;
  __bf16* wv   = wsb + OFF_WV;
  __bf16* wr   = wsb + OFF_WR;
  __bf16* wfc  = wsb + OFF_WFC;
  __bf16* xb   = wsb + OFF_XB;
  __bf16* xe   = wsb + OFF_XE;
  __bf16* relb = wsb + OFF_REL;
  __bf16* qu   = wsb + OFF_QU;
  __bf16* qv   = wsb + OFF_QV;
  __bf16* kk   = wsb + OFF_K;
  __bf16* vt   = wsb + OFF_VT;
  __bf16* rr   = wsb + OFF_R;
  __bf16* ctx  = wsb + OFF_CTX;

  k_cvt<<<HID * HID / 1024, 256, 0, stream>>>(Wq,  wq,  HID * HID);
  k_cvt<<<HID * HID / 1024, 256, 0, stream>>>(Wk,  wk,  HID * HID);
  k_cvt<<<HID * HID / 1024, 256, 0, stream>>>(Wv,  wv,  HID * HID);
  k_cvt<<<HID * HID / 1024, 256, 0, stream>>>(Wr,  wr,  HID * HID);
  k_cvt<<<HID * HID / 1024, 256, 0, stream>>>(Wfc, wfc, HID * HID);
  k_cvt<<<BB * SS * HID / 1024, 256, 0, stream>>>(x,   xb,   BB * SS * HID);
  k_cvt<<<KVN * HID / 1024,     256, 0, stream>>>(rel, relb, KVN * HID);
  k_build_xe<<<BB * KVN * HID / 1024, 256, 0, stream>>>(pkv, x, xe);

  k_proj<<<TPROJ / 8, 256, 0, stream>>>(xb, xe, relb, wq, wk, wv, wr,
                                        u, vrel, qu, qv, kk, vt, rr);

  size_t lds_bytes = ((size_t)BDROWS * KVN + 2 * QT * DD + 4 * 16 * KVT) *
                     sizeof(__bf16);  // 290816 B < 320 KB
  k_attn<<<BB * HH * (SS / QT), 128, lds_bytes, stream>>>(qu, qv, kk, vt, rr,
                                                          mask, ctx);

  k_fc<<<TFC / 8, 256, 0, stream>>>(ctx, wfc, bfc, x, out);
}
